// Generator_9749575762119
// MI455X (gfx1250) — matrix operand order repaired and checked by reading, not compiled, not run on hardware
//
#include <hip/hip_runtime.h>

// ---------------------------------------------------------------------------
// GraphRNN generator for MI455X (gfx1250, wave32, WMMA).
// Strategy: batch-parallel persistent kernel. 32 blocks x 256 threads (8
// waves). Each block owns a 16-row batch tile; all recurrent state lives in
// LDS; all GEMMs are v_wmma_f32_16x16x32_f16 with f32 accumulate. Weights are
// pre-converted (per launch, on-stream) into a WMMA-B tile layout in d_ws.
// ---------------------------------------------------------------------------

typedef _Float16 v8h  __attribute__((ext_vector_type(8)));
typedef _Float16 v16h __attribute__((ext_vector_type(16)));
typedef float    v8f  __attribute__((ext_vector_type(8)));

static constexpr int kNNode = 100;
static constexpr int kMPrev = 25;
static constexpr float kTau = 1.0f;
static constexpr int kThreads = 256;   // 8 wave32 waves
static constexpr int kWaves = 8;

// f16-element offsets of each weight matrix inside d_ws (B-operand format).
static constexpr int OFF_IH[4]  = {0, 12288, 61440, 110592};        // rnn w_ih (l0: K=25->32, N=384; l1-3: K=128)
static constexpr int OFF_HH[4]  = {159744, 208896, 258048, 307200}; // rnn w_hh (K=128, N=384)
static constexpr int OFF_MLP1   = 356352;                           // rnn_out_w1 (K=128, N=64)
static constexpr int OFF_MLP2   = 364544;                           // rnn_out_w2 (K=64, N=16)
static constexpr int OFF_ZW     = 365568;                           // z_w (K=64, N=128)
static constexpr int OFF_EIH[4] = {373760, 375296, 376832, 378368}; // out w_ih (K<=16->32, N=48)
static constexpr int OFF_EHH[4] = {379904, 381440, 382976, 384512}; // out w_hh (K=16->32, N=48)
static constexpr int WS_F16_ELEMS = 386048;                         // ~772 KB

// ---------------------------------------------------------------------------
// Weight conversion: f32 (K,N) or transposed (N,K) source -> f16 WMMA-B tiles.
// Tile (kt,nt) is 512 f16 at ((kt*Ntiles)+nt)*512; element (k,n) within tile
// at (n&15)*32 + (k&31)  => per-lane fragment = two contiguous 16B loads.
// ---------------------------------------------------------------------------
__global__ void convert_b_kernel(const float* __restrict__ src,
                                 _Float16* __restrict__ dst,
                                 int K, int N, int Kpad, int Npad, int trans) {
  int idx = blockIdx.x * 256 + threadIdx.x;
  int total = Kpad * Npad;
  if (idx >= total) return;
  int n = idx / Kpad;
  int k = idx - n * Kpad;
  float v = 0.0f;
  if (k < K && n < N) v = trans ? src[n * K + k] : src[k * N + n];
  int off = ((k >> 5) * (Npad >> 4) + (n >> 4)) * 512 + (n & 15) * 32 + (k & 31);
  dst[off] = (_Float16)v;
}

// ---------------------------------------------------------------------------
// Device helpers
// ---------------------------------------------------------------------------
__device__ __forceinline__ float sigf(float x) {
  return 1.0f / (1.0f + __expf(-x));
}

__device__ __forceinline__ float gumbel_hash(unsigned idx) {
  unsigned long long x = (unsigned long long)idx * 0x9E3779B97F4A7C15ull
                         + 0xBF58476D1CE4E5B9ull;
  x ^= x >> 30; x *= 0xBF58476D1CE4E5B9ull;
  x ^= x >> 27; x *= 0x94D049BB133111EBull;
  x ^= x >> 31;
  float u = (float)((x >> 40) & 0xFFFFFFull) * (1.0f / 16777216.0f) + 1e-7f;
  return -__logf(-__logf(u));
}

// C(16xN) = A(16xKpad f16, LDS, row-major) x B(tiled f16, global), f32 accum.
// n-tiles are distributed round-robin across the 8 waves of the block.
__device__ __forceinline__ void gemm_tiles(const _Float16* __restrict__ A,
                                           int strideA, int kts,
                                           const _Float16* __restrict__ B,
                                           int ntiles,
                                           float* __restrict__ G, int strideG) {
  const int lane = threadIdx.x & 31;
  const int wave = threadIdx.x >> 5;
  const int ln = lane & 15;   // A: row (M), B/C: col (N)
  const int hf = lane >> 4;   // half-wave selector
  for (int nt = wave; nt < ntiles; nt += kWaves) {
    v8f acc = {};
    for (int kt = 0; kt < kts; ++kt) {
      const _Float16* ap = A + ln * strideA + kt * 32 + hf * 16;
      v8h a0 = *(const v8h*)ap;
      v8h a1 = *(const v8h*)(ap + 8);
      v16h av = __builtin_shufflevector(a0, a1, 0, 1, 2, 3, 4, 5, 6, 7,
                                        8, 9, 10, 11, 12, 13, 14, 15);
      const _Float16* bp = B + (size_t)(kt * ntiles + nt) * 512 + ln * 32 + hf * 16;
      v8h b0 = *(const v8h*)bp;
      v8h b1 = *(const v8h*)(bp + 8);
      v16h bv = __builtin_shufflevector(b0, b1, 0, 1, 2, 3, 4, 5, 6, 7,
                                        8, 9, 10, 11, 12, 13, 14, 15);
      acc = __builtin_amdgcn_wmma_f32_16x16x32_f16(
          false, av, false, bv, (short)0, acc, false, false);
    }
#pragma unroll
    for (int r = 0; r < 8; ++r)
      G[(hf * 8 + r) * strideG + nt * 16 + ln] = acc[r];
  }
}

// GRU gate fusion for the node-level GRU (H=128, gates stride 384).
__device__ __forceinline__ void node_gate(const float* __restrict__ Gi,
                                          const float* __restrict__ Gh,
                                          const float* __restrict__ bi,
                                          const float* __restrict__ bh,
                                          float* __restrict__ h,
                                          _Float16* __restrict__ nextA) {
  for (int idx = threadIdx.x; idx < 16 * 128; idx += kThreads) {
    int rw = idx >> 7, c = idx & 127;
    const float* gi = Gi + rw * 384;
    const float* gh = Gh + rw * 384;
    float r = sigf(gi[c] + bi[c] + gh[c] + bh[c]);
    float z = sigf(gi[c + 128] + bi[c + 128] + gh[c + 128] + bh[c + 128]);
    float n = tanhf(gi[c + 256] + bi[c + 256] + r * (gh[c + 256] + bh[c + 256]));
    float hv = (1.0f - z) * n + z * h[idx];
    h[idx] = hv;
    nextA[idx] = (_Float16)hv;   // staged f16 input for next layer
  }
}

// GRU gate fusion for the edge-level GRU (H=16, gates stride 48); also stages
// the next layer's hidden state as the f16 A operand.
__device__ __forceinline__ void edge_gate(const float* __restrict__ Gi,
                                          const float* __restrict__ Gh,
                                          const float* __restrict__ bi,
                                          const float* __restrict__ bh,
                                          float* __restrict__ h,
                                          _Float16* __restrict__ nextA,
                                          const float* __restrict__ nexth,
                                          _Float16* __restrict__ hA) {
  int t = threadIdx.x;           // exactly 256 elements, one per thread
  int r = t >> 4, c = t & 15;
  float rg = sigf(Gi[r * 48 + c] + bi[c] + Gh[r * 48 + c] + bh[c]);
  float zg = sigf(Gi[r * 48 + 16 + c] + bi[16 + c] +
                  Gh[r * 48 + 16 + c] + bh[16 + c]);
  float ng = tanhf(Gi[r * 48 + 32 + c] + bi[32 + c] +
                   rg * (Gh[r * 48 + 32 + c] + bh[32 + c]));
  float hv = (1.0f - zg) * ng + zg * h[t];
  h[t] = hv;
  nextA[r * 32 + c] = (_Float16)hv;
  if (nexth) hA[r * 32 + c] = (_Float16)nexth[t];
}

// ---------------------------------------------------------------------------
// Main persistent kernel
// ---------------------------------------------------------------------------
__global__ __launch_bounds__(kThreads)
void generator_kernel(const float* __restrict__ Z,
                      const float* __restrict__ b_ih,
                      const float* __restrict__ b_hh,
                      const float* __restrict__ ob1,
                      const float* __restrict__ ob2,
                      const float* __restrict__ z_b,
                      const float* __restrict__ e_w,
                      const float* __restrict__ e_b,
                      const float* __restrict__ o_bih,
                      const float* __restrict__ o_bhh,
                      const float* __restrict__ oo_w1,
                      const float* __restrict__ oo_b1,
                      const float* __restrict__ oo_w2,
                      const float* __restrict__ oo_b2,
                      const _Float16* __restrict__ ws,
                      float* __restrict__ out) {
  // ---- LDS-resident state (~110 KB per workgroup; 320 KB available) ----
  __shared__ __align__(16) float s_hid[4][16 * 128];   // node GRU hidden
  __shared__ __align__(16) float s_gi[16 * 384];       // node gates (input path / MLP scratch)
  __shared__ __align__(16) float s_gh[16 * 384];       // node gates (hidden path)
  __shared__ __align__(16) float s_h[16 * 16];         // node MLP output -> edge init
  __shared__ __align__(16) float s_ehid[4][16 * 16];   // edge GRU hidden
  __shared__ __align__(16) float s_egi[16 * 48];
  __shared__ __align__(16) float s_egh[16 * 48];
  __shared__ __align__(16) float s_x[16 * 32];         // x_step rows (25 used, padded to 32)
  __shared__ __align__(16) float s_y1[16 * 8];
  __shared__             float s_ox[16];
  __shared__ __align__(16) _Float16 s_inA[16 * 128];   // staged layer input (f16 A)
  __shared__ __align__(16) _Float16 s_hA[16 * 128];    // staged hidden (f16 A)
  __shared__ __align__(16) _Float16 s_xA[16 * 32];
  __shared__ __align__(16) _Float16 s_eA[16 * 32];
  __shared__ __align__(16) _Float16 s_eInA[16 * 32];
  __shared__ __align__(16) _Float16 s_eHA[16 * 32];
  __shared__ __align__(16) _Float16 s_zA[16 * 64];

  const int tid = threadIdx.x;
  const int brow0 = blockIdx.x * 16;

  // ---- init: zero f16 pads, zero hidden layers 1..3, stage Z, x = ones ----
  for (int t = tid; t < 16 * 32; t += kThreads) {
    s_eA[t] = (_Float16)0.f; s_eInA[t] = (_Float16)0.f;
    s_eHA[t] = (_Float16)0.f; s_xA[t] = (_Float16)0.f;
    s_x[t] = ((t & 31) < kMPrev) ? 1.0f : 0.0f;
  }
  for (int t = tid; t < 16 * 128; t += kThreads) {
    s_hid[1][t] = 0.f; s_hid[2][t] = 0.f; s_hid[3][t] = 0.f;
  }
  for (int t = tid; t < 16 * 64; t += kThreads) {
    int r = t >> 6, c = t & 63;
    s_zA[t] = (_Float16)Z[(brow0 + r) * 64 + c];
  }
  __syncthreads();

  // hid[0] = Z @ z_w + z_b  (WMMA: K=64 -> 2 ktiles, N=128 -> 8 ntiles)
  gemm_tiles(s_zA, 64, 2, ws + OFF_ZW, 8, s_hid[0], 128);
  __syncthreads();
  for (int t = tid; t < 16 * 128; t += kThreads) s_hid[0][t] += z_b[t & 127];
  __syncthreads();

  // ---- node-level step: 4-layer GRU stack + output MLP (all WMMA) ----
  auto node_step = [&]() {
    for (int l = 0; l < 4; ++l) {
      for (int t = tid; t < 16 * 128; t += kThreads)
        s_hA[t] = (_Float16)s_hid[l][t];
      if (l == 0)
        for (int t = tid; t < 16 * 32; t += kThreads)
          s_xA[t] = (_Float16)s_x[t];
      __syncthreads();
      if (l == 0)
        gemm_tiles(s_xA, 32, 1, ws + OFF_IH[0], 24, s_gi, 384);
      else
        gemm_tiles(s_inA, 128, 4, ws + OFF_IH[l], 24, s_gi, 384);
      gemm_tiles(s_hA, 128, 4, ws + OFF_HH[l], 24, s_gh, 384);
      __syncthreads();
      node_gate(s_gi, s_gh, b_ih + l * 384, b_hh + l * 384, s_hid[l], s_inA);
      __syncthreads();
    }
    // h = relu(top @ w1 + b1) @ w2 + b2
    gemm_tiles(s_inA, 128, 4, ws + OFF_MLP1, 4, s_gi, 64);
    __syncthreads();
    for (int t = tid; t < 16 * 64; t += kThreads) {
      float v = s_gi[t] + ob1[t & 63];
      s_hA[t] = (_Float16)fmaxf(v, 0.f);
    }
    __syncthreads();
    gemm_tiles(s_hA, 64, 2, ws + OFF_MLP2, 1, s_h, 16);
    __syncthreads();
    for (int t = tid; t < 256; t += kThreads) s_h[t] += ob2[t & 15];
    __syncthreads();
  };

  node_step();   // initial step with x = ones

  for (int i = 0; i < kNNode; ++i) {
    // edge hidden: layer0 = h, rest 0; ox = 1
    {
      s_ehid[0][tid] = s_h[tid];
      s_ehid[1][tid] = 0.f; s_ehid[2][tid] = 0.f; s_ehid[3][tid] = 0.f;
      if (tid < 16) s_ox[tid] = 1.0f;
    }
    __syncthreads();

    for (int j = 0; j < kMPrev; ++j) {
      // emb = relu(ox * emb_w + emb_b); stage emb + layer0 hidden as f16 A
      if (tid < 128) {
        int r = tid >> 3, c = tid & 7;
        s_eA[r * 32 + c] = (_Float16)fmaxf(s_ox[r] * e_w[c] + e_b[c], 0.f);
      }
      {
        int r = tid >> 4, c = tid & 15;
        s_eHA[r * 32 + c] = (_Float16)s_ehid[0][tid];
      }
      __syncthreads();
      for (int l = 0; l < 4; ++l) {
        gemm_tiles(l == 0 ? s_eA : s_eInA, 32, 1, ws + OFF_EIH[l], 3, s_egi, 48);
        gemm_tiles(s_eHA, 32, 1, ws + OFF_EHH[l], 3, s_egh, 48);
        __syncthreads();
        edge_gate(s_egi, s_egh, o_bih + l * 48, o_bhh + l * 48, s_ehid[l],
                  s_eInA, (l < 3) ? s_ehid[l + 1] : nullptr, s_eHA);
        __syncthreads();
      }
      // tiny MLP head + log-sigmoid + Gumbel-softmax (VALU; K too small for WMMA)
      if (tid < 128) {
        int r = tid >> 3, c = tid & 7;
        float acc = oo_b1[c];
#pragma unroll
        for (int k = 0; k < 16; ++k) acc += s_ehid[3][r * 16 + k] * oo_w1[k * 8 + c];
        s_y1[tid] = fmaxf(acc, 0.f);
      }
      __syncthreads();
      if (tid < 16) {
        float y = oo_b2[0];
#pragma unroll
        for (int c = 0; c < 8; ++c) y += s_y1[tid * 8 + c] * oo_w2[c];
        float logp = fminf(y, 0.f) - log1pf(__expf(-fabsf(y)));
        unsigned base = (((unsigned)i * 25u + (unsigned)j) * 512u
                         + (unsigned)(brow0 + tid)) * 2u;
        float g0 = gumbel_hash(base);
        float g1 = gumbel_hash(base + 1u);
        // softmax over [logp+g0, -logp+g1] row 0  ==  sigmoid(2*logp+g0-g1)
        float ox2 = sigf((2.0f * logp + g0 - g1) * (1.0f / kTau));
        s_ox[tid] = ox2;
        s_x[tid * 32 + j] = (j <= i) ? ox2 : 0.0f;
      }
      __syncthreads();
    }

    // emit this node's row of the adjacency output: (B, N_NODE, M_PREV)
    for (int t = tid; t < 16 * kMPrev; t += kThreads) {
      int r = t / kMPrev, m = t - r * kMPrev;
      out[(size_t)(brow0 + r) * (kNNode * kMPrev) + (size_t)i * kMPrev + m] =
          s_x[r * 32 + m];
    }
    __syncthreads();

    node_step();   // node-level GRU update with x_step
  }
}

// ---------------------------------------------------------------------------
// Host entry
// ---------------------------------------------------------------------------
extern "C" void kernel_launch(void* const* d_in, const int* in_sizes, int n_in,
                              void* d_out, int out_size, void* d_ws,
                              size_t ws_size, hipStream_t stream) {
  (void)in_sizes; (void)n_in; (void)out_size; (void)ws_size;

  const float* Z     = (const float*)d_in[0];
  const float* w_ih0 = (const float*)d_in[1];
  const float* w_ihL = (const float*)d_in[2];
  const float* w_hh  = (const float*)d_in[3];
  const float* b_ih  = (const float*)d_in[4];
  const float* b_hh  = (const float*)d_in[5];
  const float* ow1   = (const float*)d_in[6];
  const float* ob1   = (const float*)d_in[7];
  const float* ow2   = (const float*)d_in[8];
  const float* ob2   = (const float*)d_in[9];
  const float* z_w   = (const float*)d_in[10];
  const float* z_b   = (const float*)d_in[11];
  const float* e_w   = (const float*)d_in[12];
  const float* e_b   = (const float*)d_in[13];
  const float* o_ih0 = (const float*)d_in[14];
  const float* o_ihL = (const float*)d_in[15];
  const float* o_hh  = (const float*)d_in[16];
  const float* o_bih = (const float*)d_in[17];
  const float* o_bhh = (const float*)d_in[18];
  const float* oo_w1 = (const float*)d_in[19];
  const float* oo_b1 = (const float*)d_in[20];
  const float* oo_w2 = (const float*)d_in[21];
  const float* oo_b2 = (const float*)d_in[22];
  _Float16* ws = (_Float16*)d_ws;
  float* out = (float*)d_out;

  auto cv = [&](const float* src, int off, int K, int N, int trans) {
    int Kpad = (K + 31) & ~31, Npad = (N + 15) & ~15;
    int total = Kpad * Npad;
    convert_b_kernel<<<(total + 255) / 256, 256, 0, stream>>>(
        src, ws + off, K, N, Kpad, Npad, trans);
  };

  // GEMM B operands: "trans=1" means B = W^T (x @ W.T); "trans=0" means B = W.
  cv(w_ih0, OFF_IH[0], 25, 384, 1);
  cv(w_ihL + 0 * 384 * 128, OFF_IH[1], 128, 384, 1);
  cv(w_ihL + 1 * 384 * 128, OFF_IH[2], 128, 384, 1);
  cv(w_ihL + 2 * 384 * 128, OFF_IH[3], 128, 384, 1);
  for (int l = 0; l < 4; ++l) cv(w_hh + l * 384 * 128, OFF_HH[l], 128, 384, 1);
  cv(ow1, OFF_MLP1, 128, 64, 0);
  cv(ow2, OFF_MLP2, 64, 16, 0);
  cv(z_w, OFF_ZW, 64, 128, 0);
  cv(o_ih0, OFF_EIH[0], 8, 48, 1);
  for (int l = 1; l < 4; ++l) cv(o_ihL + (l - 1) * 48 * 16, OFF_EIH[l], 16, 48, 1);
  for (int l = 0; l < 4; ++l) cv(o_hh + l * 48 * 16, OFF_EHH[l], 16, 48, 1);

  generator_kernel<<<32, kThreads, 0, stream>>>(
      Z, b_ih, b_hh, ob1, ob2, z_b, e_w, e_b, o_bih, o_bhh,
      oo_w1, oo_b1, oo_w2, oo_b2, ws, out);
}
